// M_Metric_test_49271864820308
// MI455X (gfx1250) — matrix operand order repaired and checked by reading, not compiled, not run on hardware
//
#include <hip/hip_runtime.h>

// ---------------------------------------------------------------------------
// p[n] = diff_n^T * M * diff_n, diff_n = X[0] - X[1+n], D = 256.
// Per 16-row tile: Y = diff @ M via v_wmma_f32_16x16x32_bf16, then
// p = rowsum(Y .* diff). HBM-bound (~268 MB streamed, ~11.5 us floor).
// All 8 A-fragments preloaded per tile; 16 N-blocks done in two halves of
// 8 C-tiles; B fragments double-buffered. Target: zero scratch spills.
// ---------------------------------------------------------------------------

typedef __attribute__((ext_vector_type(16))) __bf16 v16bf;
typedef __attribute__((ext_vector_type(8)))  float  v8f;

#define D_DIM 256
#define ROW_PITCH 264                 // bf16 elems per diff row (256 + 8 pad) = 528 B
#define WAVES_PER_BLOCK 8
#define BLOCK_THREADS (WAVES_PER_BLOCK * 32)
#define NBLOCKS 256

// Dynamic LDS layout:
//   [0,            131072)  Mpk  : 128 frags * 32 lanes * 16 bf16 (B-fragment order)
//   [131072,       132096)  row0 : 256 f32
//   [132096, ...)           diff : per-wave 16 rows * 528 B
#define MPK_BYTES       (128 * 32 * 16 * 2)
#define ROW0_OFF        MPK_BYTES
#define ROW0_BYTES      (D_DIM * 4)
#define DIFF_OFF        (ROW0_OFF + ROW0_BYTES)
#define DIFF_WAVE_BYTES (16 * ROW_PITCH * 2)
#define SMEM_BYTES      (DIFF_OFF + WAVES_PER_BLOCK * DIFF_WAVE_BYTES)

__device__ __forceinline__ unsigned pack_bf2(float lo, float hi) {
    unsigned short l = __builtin_bit_cast(unsigned short, (__bf16)lo);
    unsigned short h = __builtin_bit_cast(unsigned short, (__bf16)hi);
    return (unsigned)l | ((unsigned)h << 16);
}

__device__ __forceinline__ v16bf load_frag2(const __bf16* p0, const __bf16* p1) {
    v16bf v;
    ((uint4*)&v)[0] = *(const uint4*)p0;
    ((uint4*)&v)[1] = *(const uint4*)p1;
    return v;
}

__global__ __launch_bounds__(BLOCK_THREADS, 1)
void mmetric_wmma_kernel(const float* __restrict__ X,
                         const float* __restrict__ M,
                         float* __restrict__ out,
                         int nrows) {
    extern __shared__ char smem[];
    __bf16* Mpk  = (__bf16*)smem;
    float*  row0 = (float*)(smem + ROW0_OFF);

    const int tid  = threadIdx.x;
    const int lane = tid & 31;
    const int wid  = tid >> 5;
    const int g    = lane >> 4;   // lane half (0: lanes 0-15, 1: lanes 16-31)
    const int ln   = lane & 15;

    __bf16* diffw = (__bf16*)(smem + DIFF_OFF + wid * DIFF_WAVE_BYTES);

    // ---- Stage row 0 (fp32) ----
    for (int i = tid; i < D_DIM; i += BLOCK_THREADS) row0[i] = X[i];

    // ---- Stage M as bf16 in pre-swizzled WMMA B-fragment order ----
    // B frag (kb,nb): lane l, elem e8 holds M[kb*32 + (l>>4)*16 + e8][nb*16 + (l&15)]
    for (int t = tid; t < D_DIM * D_DIM; t += BLOCK_THREADS) {
        const int d = t >> 8;
        const int e = t & 255;
        const float m = M[t];
        const int idx = (((d >> 5) * 16 + (e >> 4)) * 32 +
                         ((d >> 4) & 1) * 16 + (e & 15)) * 16 + (d & 15);
        Mpk[idx] = (__bf16)m;
    }
    __syncthreads();

    const int tiles      = (nrows + 15) >> 4;
    const int wave_glob  = blockIdx.x * WAVES_PER_BLOCK + wid;
    const int wave_count = gridDim.x * WAVES_PER_BLOCK;

    for (int tile = wave_glob; tile < tiles; tile += wave_count) {
        // ---- 1) Build 16x256 bf16 diff tile in LDS ----
        // Row-0 slice read per tile (short live range; LDS is cheap).
        float a[8];
        {
            const float4* r = (const float4*)(row0 + lane * 8);
            const float4 a0 = r[0], a1 = r[1];
            a[0] = a0.x; a[1] = a0.y; a[2] = a0.z; a[3] = a0.w;
            a[4] = a1.x; a[5] = a1.y; a[6] = a1.z; a[7] = a1.w;
        }
        const float* xbase = X + (size_t)(1 + tile * 16) * D_DIM;
        const bool   pf_ok = (tile + wave_count) < tiles;
        const float* nbase = X + (size_t)(1 + (size_t)(tile + wave_count) * 16) * D_DIM;
#pragma unroll 4
        for (int r = 0; r < 16; ++r) {
            const int ri = tile * 16 + r;
            uint4 w = {0u, 0u, 0u, 0u};
            if (ri < nrows) {
                const float4* p  = (const float4*)(xbase + (size_t)r * D_DIM + lane * 8);
                const float4  x0 = p[0];
                const float4  x1 = p[1];
                w.x = pack_bf2(a[0] - x0.x, a[1] - x0.y);
                w.y = pack_bf2(a[2] - x0.z, a[3] - x0.w);
                w.z = pack_bf2(a[4] - x1.x, a[5] - x1.y);
                w.w = pack_bf2(a[6] - x1.z, a[7] - x1.w);
            }
            *(uint4*)(diffw + r * ROW_PITCH + lane * 8) = w;
            if (pf_ok)  // pull next tile into GL2 while we compute (global_prefetch_b8)
                __builtin_prefetch(nbase + (size_t)r * D_DIM + lane * 8, 0, 0);
        }

        // ---- 2) Preload all 8 A fragments (16x32 bf16 each) for this tile ----
        // Lane half g -> K = kb*32 + g*16 + {0..15}; row = ln.
        v16bf Afrag[8];
        {
            const __bf16* arow = diffw + ln * ROW_PITCH + g * 16;
#pragma unroll
            for (int kb = 0; kb < 8; ++kb)
                Afrag[kb] = load_frag2(arow + kb * 32, arow + kb * 32 + 8);
        }

        float acc[8] = {0.f, 0.f, 0.f, 0.f, 0.f, 0.f, 0.f, 0.f};

        // ---- 3) Two halves of 8 N-blocks: 8 live C-tiles each ----
#pragma unroll
        for (int half = 0; half < 2; ++half) {
            const int nb0 = half * 8;
            v8f C[8];
            const v8f vzero = {0.f, 0.f, 0.f, 0.f, 0.f, 0.f, 0.f, 0.f};
#pragma unroll
            for (int i = 0; i < 8; ++i) C[i] = vzero;

#pragma unroll
            for (int kb = 0; kb < 8; ++kb) {
                // B fragments double-buffered to overlap LDS with the WMMA pipe.
                const __bf16* bbase = Mpk + ((kb * 16 + nb0) * 32 + lane) * 16;
                v16bf Bcur = load_frag2(bbase, bbase + 8);
#pragma unroll
                for (int nb8 = 0; nb8 < 8; ++nb8) {
                    v16bf Bnext = Bcur;
                    if (nb8 < 7) {
                        const __bf16* bp = bbase + (nb8 + 1) * (32 * 16);
                        Bnext = load_frag2(bp, bp + 8);
                    }
                    C[nb8] = __builtin_amdgcn_wmma_f32_16x16x32_bf16(
                        false, Afrag[kb], false, Bcur, (short)0, C[nb8], false, false);
                    Bcur = Bnext;
                }
            }

            // ---- 4) p partials for this half: C .* diff in C/D VGPR layout ----
            // VGPR r, lane half g -> row r + 8g, col (nb0+nb8)*16 + ln
#pragma unroll
            for (int nb8 = 0; nb8 < 8; ++nb8) {
                const __bf16* crow = diffw + (nb0 + nb8) * 16 + ln;
#pragma unroll
                for (int r = 0; r < 8; ++r)
                    acc[r] += C[nb8][r] * (float)crow[(r + 8 * g) * ROW_PITCH];
            }
        }

        // ---- 5) Reduce across the 16 lanes of each half; lanes 0/16 store ----
#pragma unroll
        for (int r = 0; r < 8; ++r) {
            float v = acc[r];
            v += __shfl_xor(v, 1, 16);
            v += __shfl_xor(v, 2, 16);
            v += __shfl_xor(v, 4, 16);
            v += __shfl_xor(v, 8, 16);
            acc[r] = v;
        }
        if (ln == 0) {
            const int base = tile * 16 + g * 8;
#pragma unroll
            for (int r = 0; r < 8; ++r)
                if (base + r < nrows) out[base + r] = acc[r];
        }
    }
}

extern "C" void kernel_launch(void* const* d_in, const int* in_sizes, int n_in,
                              void* d_out, int out_size, void* d_ws, size_t ws_size,
                              hipStream_t stream) {
    const float* X = (const float*)d_in[0];   // [N, 256] fp32
    const float* M = (const float*)d_in[1];   // [256, 256] fp32
    float* out = (float*)d_out;               // [N-1] fp32

    const int N     = in_sizes[0] / D_DIM;
    const int nrows = N - 1;

    (void)hipFuncSetAttribute(reinterpret_cast<const void*>(mmetric_wmma_kernel),
                              hipFuncAttributeMaxDynamicSharedMemorySize, SMEM_BYTES);

    const int tiles     = (nrows + 15) / 16;
    const int maxBlocks = (tiles + WAVES_PER_BLOCK - 1) / WAVES_PER_BLOCK;
    const int blocks    = (NBLOCKS < maxBlocks) ? NBLOCKS : maxBlocks;

    mmetric_wmma_kernel<<<blocks, BLOCK_THREADS, SMEM_BYTES, stream>>>(X, M, out, nrows);
}
